// SparseMoELayer_81922206204359
// MI455X (gfx1250) — compile-verified, matrix operand order checked by reading, not run on hardware
//
#include <hip/hip_runtime.h>
#include <math.h>

// ---------------- problem constants ----------------
#define H_DIM 1024
#define FF_DIM 4096
#define N_TOK 16384
#define N_EXP 8
#define TOPK 2
#define NK (N_TOK * TOPK)

// ---------------- WMMA types (gfx1250, wave32) ----------------
typedef __bf16 bf16;
typedef __attribute__((ext_vector_type(16))) __bf16 v16bf;
typedef __attribute__((ext_vector_type(8)))  __bf16 v8bf;
typedef __attribute__((ext_vector_type(8)))  float  v8f;

// A/B fragment (16x32 / 32x16 bf16): per lane two contiguous 8-element (16B)
// chunks at K = base and base+16 (half-wave selects +8 inside each 16-chunk).
__device__ __forceinline__ v16bf ld_frag16(const bf16* p_lo, const bf16* p_hi) {
  v8bf lo = *(const v8bf*)p_lo;
  v8bf hi = *(const v8bf*)p_hi;
  v16bf r;
#pragma unroll
  for (int i = 0; i < 8; ++i) { r[i] = lo[i]; r[i + 8] = hi[i]; }
  return r;
}

// ---------------- stage 0: re-init workspace counters ----------------
__global__ void moe_init(int* __restrict__ counts) {
  if (threadIdx.x < N_EXP) counts[threadIdx.x] = 0;
}

// ---------------- stage 1: router (one wave32 per token) ----------------
__global__ void moe_router(const float* __restrict__ x, const float* __restrict__ grad,
                           const float* __restrict__ rw, const float* __restrict__ rb,
                           float* __restrict__ out, float* __restrict__ probs,
                           int* __restrict__ counts, int* __restrict__ tok_expert,
                           float* __restrict__ tok_weight) {
  const int wave = threadIdx.x >> 5;
  const int lane = threadIdx.x & 31;
  const int t = blockIdx.x * 8 + wave;

  const float* xr = x + (size_t)t * H_DIM;
  float acc[N_EXP];
#pragma unroll
  for (int e = 0; e < N_EXP; ++e) acc[e] = 0.0f;
  for (int h = lane; h < H_DIM; h += 32) {
    const float xv = xr[h];
    const float* w = rw + (size_t)h * N_EXP;
#pragma unroll
    for (int e = 0; e < N_EXP; ++e) acc[e] += xv * w[e];
  }
#pragma unroll
  for (int e = 0; e < N_EXP; ++e) {
#pragma unroll
    for (int off = 16; off > 0; off >>= 1) acc[e] += __shfl_xor(acc[e], off, 32);
  }

  // zero the output row; FFN kernel accumulates atomically later (stream order)
  float4* orow = (float4*)(out + (size_t)t * H_DIM);
  float4 z; z.x = z.y = z.z = z.w = 0.0f;
  for (int j = lane; j < H_DIM / 4; j += 32) orow[j] = z;

  if (lane == 0) {
    const float g = grad[t];
    float p[N_EXP];
    float m = -1e30f;
#pragma unroll
    for (int e = 0; e < N_EXP; ++e) {
      p[e] = acc[e] + g * rw[(size_t)H_DIM * N_EXP + e] + rb[e];
      m = fmaxf(m, p[e]);
    }
    float s = 0.0f;
#pragma unroll
    for (int e = 0; e < N_EXP; ++e) { p[e] = expf(p[e] - m); s += p[e]; }
    const float inv = 1.0f / s;
#pragma unroll
    for (int e = 0; e < N_EXP; ++e) { p[e] *= inv; probs[(size_t)t * N_EXP + e] = p[e]; }

    int i1 = 0;
#pragma unroll
    for (int e = 1; e < N_EXP; ++e) if (p[e] > p[i1]) i1 = e;
    int i2 = (i1 == 0) ? 1 : 0;
#pragma unroll
    for (int e = 0; e < N_EXP; ++e) if (e != i1 && p[e] > p[i2]) i2 = e;

    tok_expert[2 * t + 0] = i1; tok_weight[2 * t + 0] = p[i1];
    tok_expert[2 * t + 1] = i2; tok_weight[2 * t + 1] = p[i2];
    atomicAdd(&counts[i1], 1);
    atomicAdd(&counts[i2], 1);
  }
}

// ---------------- stage 2: exclusive scan of expert counts ----------------
__global__ void moe_scan(const int* __restrict__ counts, int* __restrict__ offsets,
                         int* __restrict__ cursor) {
  if (threadIdx.x == 0) {
    int s = 0;
#pragma unroll
    for (int e = 0; e < N_EXP; ++e) { offsets[e] = s; cursor[e] = s; s += counts[e]; }
    offsets[N_EXP] = s;
  }
}

// ---------------- stage 3: scatter token slots into per-expert lists ----------------
__global__ void moe_scatter(const int* __restrict__ tok_expert,
                            const float* __restrict__ tok_weight,
                            int* __restrict__ cursor, int* __restrict__ erows,
                            float* __restrict__ ewts) {
  const int i = blockIdx.x * blockDim.x + threadIdx.x;
  if (i < NK) {
    const int e = tok_expert[i];
    const int pos = atomicAdd(&cursor[e], 1);
    erows[pos] = i >> 1;          // token id
    ewts[pos]  = tok_weight[i];   // routing prob
  }
}

// ---------------- stage 4: fused expert FFN (WMMA bf16) ----------------
// Tile: 32 tokens x full H; ff processed in chunks of 64; 8 waves/block.
// W1 k-chunks double-buffered: 1 barrier per k-step instead of 2.
#define XS_STR  1032   // 1024 + 8 pad (16B-aligned frag loads)
#define W1T_STR 136    // 128 + 8
#define W2T_STR 72     // 64 + 8
#define H1_STR  72     // 64 + 8
#define W1BUF   (64 * W1T_STR)
#define SMEM_BYTES (((32 * XS_STR) + (2 * W1BUF) + (1024 * W2T_STR) + (32 * H1_STR)) * 2 + 32 * 4 + 32 * 4)

// stage W1[ks:ks+128, f:f+64] -> buf[col][k] (bf16, transposed)
__device__ __forceinline__ void stage_w1(const float* __restrict__ W1e, bf16* __restrict__ buf,
                                         int ks, int f, int tid) {
  const int r0 = tid >> 4;
  const int c4 = (tid & 15) * 4;
#pragma unroll
  for (int rr = 0; rr < 8; ++rr) {
    const int r = r0 + rr * 16;
    const float4 v = *(const float4*)(W1e + (size_t)(ks + r) * FF_DIM + f + c4);
    buf[(c4 + 0) * W1T_STR + r] = (bf16)v.x;
    buf[(c4 + 1) * W1T_STR + r] = (bf16)v.y;
    buf[(c4 + 2) * W1T_STR + r] = (bf16)v.z;
    buf[(c4 + 3) * W1T_STR + r] = (bf16)v.w;
  }
}

__global__ void __launch_bounds__(256, 1) moe_ffn(
    const float* __restrict__ x,
    const float* __restrict__ w1, const float* __restrict__ b1,
    const float* __restrict__ w2, const float* __restrict__ b2,
    const int* __restrict__ offsets, const int* __restrict__ erows,
    const float* __restrict__ ewts, float* __restrict__ out) {
  extern __shared__ char smem_raw[];
  bf16* Xs   = (bf16*)smem_raw;            // [32][XS_STR]  gathered X tile (bf16)
  bf16* W1d  = Xs + 32 * XS_STR;           // [2][64][W1T_STR] W1 chunk, double-buffered
  bf16* W2sT = W1d + 2 * W1BUF;            // [1024][W2T_STR] W2 chunk, transposed
  bf16* H1s  = W2sT + 1024 * W2T_STR;      // [32][H1_STR]  gelu(h) tile
  float* wts_s = (float*)(H1s + 32 * H1_STR);
  int*   rows_s = (int*)(wts_s + 32);

  const int e = blockIdx.y;
  const int beg = offsets[e];
  const int end = offsets[e + 1];
  const int start = beg + blockIdx.x * 32;
  if (start >= end) return;                       // block-uniform early exit
  const int nrows = (end - start < 32) ? (end - start) : 32;

  const int tid  = threadIdx.x;
  const int wave = tid >> 5;
  const int lane = tid & 31;
  const int l15  = lane & 15;
  const int khalf = (lane < 16) ? 0 : 8;          // half-wave K offset (A & B frags)

  const float* W1e = w1 + (size_t)e * H_DIM * FF_DIM;
  const float* W2e = w2 + (size_t)e * FF_DIM * H_DIM;
  const float* B1e = b1 + (size_t)e * FF_DIM;
  const float* B2e = b2 + (size_t)e * H_DIM;

  if (tid < 32) {
    const bool ok = tid < nrows;
    rows_s[tid] = ok ? erows[start + tid] : 0;
    wts_s[tid]  = ok ? ewts[start + tid]  : 0.0f;
  }
  __syncthreads();

  { // gather + convert X tile: 32 rows x 1024 f32 -> bf16 LDS
    const int r  = tid >> 3;
    const int cb = (tid & 7) * 128;
    const float4* src = (const float4*)(x + (size_t)rows_s[r] * H_DIM + cb);
    bf16* dst = Xs + r * XS_STR + cb;
    const bool valid = r < nrows;
#pragma unroll
    for (int j = 0; j < 32; ++j) {
      float4 v;
      if (valid) v = src[j]; else { v.x = v.y = v.z = v.w = 0.0f; }
      dst[j * 4 + 0] = (bf16)v.x; dst[j * 4 + 1] = (bf16)v.y;
      dst[j * 4 + 2] = (bf16)v.z; dst[j * 4 + 3] = (bf16)v.w;
    }
  }

  const int m16 = wave & 1;   // GEMM1: wave's 16x16 tile of H1 [32 x 64]
  const int n16 = wave >> 1;

  // W2 staging coords (reused every ff-chunk)
  const int w2r  = tid >> 2;
  const int w2cb = (tid & 3) * 256;

  // GEMM2 accumulators: wave owns out cols [wave*128, wave*128+128): 2m x 8n tiles
  v8f c2[2][8];
#pragma unroll
  for (int mi = 0; mi < 2; ++mi)
#pragma unroll
    for (int ni = 0; ni < 8; ++ni)
#pragma unroll
      for (int r = 0; r < 8; ++r) c2[mi][ni][r] = 0.0f;

  for (int f = 0; f < FF_DIM; f += 64) {
    v8f c1;
#pragma unroll
    for (int r = 0; r < 8; ++r) c1[r] = 0.0f;

    // fence previous GEMM2 readers of W2sT/H1s (and iter-6 readers of W1 buf0)
    __syncthreads();

    { // stage W2[f:f+64, 0:1024] -> W2sT[col][k] early: latency hides under GEMM1
      const float4* src = (const float4*)(W2e + (size_t)(f + w2r) * H_DIM + w2cb);
#pragma unroll
      for (int j = 0; j < 64; ++j) {
        const float4 v = src[j];
        const int c = w2cb + j * 4;
        W2sT[(c + 0) * W2T_STR + w2r] = (bf16)v.x;
        W2sT[(c + 1) * W2T_STR + w2r] = (bf16)v.y;
        W2sT[(c + 2) * W2T_STR + w2r] = (bf16)v.z;
        W2sT[(c + 3) * W2T_STR + w2r] = (bf16)v.w;
      }
    }
    // prefetch next ff-chunk's W2 rows into L2/WGP (global_prefetch_b8)
    if (f + 64 < FF_DIM)
      __builtin_prefetch(W2e + (size_t)(f + 64 + w2r) * H_DIM + w2cb, 0, 3);

    stage_w1(W1e, W1d, 0, f, tid);    // prime buffer 0

    // -------- GEMM1: H1[32x64] = X[32x1024] @ W1[:, f:f+64] (pipelined) --------
    const bf16* arow_base = Xs + (m16 * 16 + l15) * XS_STR + khalf;
    for (int i = 0; i < 8; ++i) {
      __syncthreads();                      // stage(i) complete; readers(i-1) done
      const bf16* arow = arow_base + i * 128;
      const bf16* brow = W1d + (i & 1) * W1BUF + (n16 * 16 + l15) * W1T_STR + khalf;
#pragma unroll
      for (int kk = 0; kk < 128; kk += 32) {
        const v16bf a = ld_frag16(arow + kk, arow + kk + 16);
        const v16bf b = ld_frag16(brow + kk, brow + kk + 16);
        c1 = __builtin_amdgcn_wmma_f32_16x16x32_bf16(false, a, false, b,
                                                     (short)0, c1, false, false);
      }
      if (i < 7)                            // stage next chunk into other buffer
        stage_w1(W1e, W1d + ((i + 1) & 1) * W1BUF, (i + 1) * 128, f, tid);
    }

    { // bias + exact GELU, convert to bf16, store H1 tile
      const float bv = B1e[f + n16 * 16 + l15];
      const int mb = m16 * 16 + khalf;
#pragma unroll
      for (int r = 0; r < 8; ++r) {
        const float h = c1[r] + bv;
        const float gel = 0.5f * h * (1.0f + erff(h * 0.70710678118654752f));
        H1s[(mb + r) * H1_STR + n16 * 16 + l15] = (bf16)gel;
      }
    }
    __syncthreads();                        // H1s + W2sT ready for all waves

    // -------- GEMM2: out[32x1024] += H1[32x64] @ W2[f:f+64, :] --------
#pragma unroll
    for (int mi = 0; mi < 2; ++mi) {
      const bf16* ar = H1s + (mi * 16 + l15) * H1_STR + khalf;
#pragma unroll
      for (int kk = 0; kk < 64; kk += 32) {
        const v16bf a = ld_frag16(ar + kk, ar + kk + 16);
#pragma unroll
        for (int ni = 0; ni < 8; ++ni) {
          const bf16* br = W2sT + (wave * 128 + ni * 16 + l15) * W2T_STR + khalf;
          const v16bf b = ld_frag16(br + kk, br + kk + 16);
          c2[mi][ni] = __builtin_amdgcn_wmma_f32_16x16x32_bf16(false, a, false, b,
                                                              (short)0, c2[mi][ni],
                                                              false, false);
        }
      }
    }
  }

  // -------- epilogue: out[token] += wt * (acc + b2) (atomic: K=2 overlap) --------
  float b2v[8];
#pragma unroll
  for (int ni = 0; ni < 8; ++ni) b2v[ni] = B2e[wave * 128 + ni * 16 + l15];
#pragma unroll
  for (int mi = 0; mi < 2; ++mi) {
#pragma unroll
    for (int r = 0; r < 8; ++r) {
      const int lrow = mi * 16 + khalf + r;   // C layout: reg r -> M = r + halfwave*8
      const float wt = wts_s[lrow];
      if (wt != 0.0f) {
        float* orow = out + (size_t)rows_s[lrow] * H_DIM;
#pragma unroll
        for (int ni = 0; ni < 8; ++ni) {
          atomicAdd(&orow[wave * 128 + ni * 16 + l15], wt * (c2[mi][ni][r] + b2v[ni]));
        }
      }
    }
  }
}

// ---------------- host launch ----------------
extern "C" void kernel_launch(void* const* d_in, const int* in_sizes, int n_in,
                              void* d_out, int out_size, void* d_ws, size_t ws_size,
                              hipStream_t stream) {
  (void)in_sizes; (void)n_in; (void)out_size; (void)ws_size;
  const float* x    = (const float*)d_in[0];
  const float* grad = (const float*)d_in[1];
  const float* rw   = (const float*)d_in[2];
  const float* rb   = (const float*)d_in[3];
  const float* w1   = (const float*)d_in[4];
  const float* b1   = (const float*)d_in[5];
  const float* w2   = (const float*)d_in[6];
  const float* b2   = (const float*)d_in[7];

  float* out   = (float*)d_out;
  float* probs = out + (size_t)N_TOK * H_DIM;   // tuple output #2

  // workspace layout (~512 KB)
  int*   counts     = (int*)d_ws;               // 8
  int*   offsets    = counts + 8;               // 9
  int*   cursor     = offsets + 9;              // 8 (meta padded to 32 ints)
  int*   tok_expert = (int*)d_ws + 32;          // NK
  float* tok_weight = (float*)(tok_expert + NK);// NK
  int*   erows      = (int*)(tok_weight + NK);  // NK
  float* ewts       = (float*)(erows + NK);     // NK

  moe_init<<<1, 64, 0, stream>>>(counts);
  moe_router<<<N_TOK / 8, 256, 0, stream>>>(x, grad, rw, rb, out, probs,
                                            counts, tok_expert, tok_weight);
  moe_scan<<<1, 32, 0, stream>>>(counts, offsets, cursor);
  moe_scatter<<<NK / 256, 256, 0, stream>>>(tok_expert, tok_weight, cursor, erows, ewts);

  static_assert(SMEM_BYTES <= 320 * 1024, "LDS budget exceeded");
  hipFuncSetAttribute((const void*)moe_ffn,
                      hipFuncAttributeMaxDynamicSharedMemorySize, SMEM_BYTES);
  moe_ffn<<<dim3(N_TOK / 32, N_EXP), 256, SMEM_BYTES, stream>>>(
      x, w1, b1, w2, b2, offsets, erows, ewts, out);
}
